// AttentionBlock_64166811402695
// MI455X (gfx1250) — matrix operand order repaired and checked by reading, not compiled, not run on hardware
//
#include <hip/hip_runtime.h>

typedef __attribute__((ext_vector_type(16))) _Float16     v16h;
typedef __attribute__((ext_vector_type(8)))  float        v8f;
typedef __attribute__((ext_vector_type(4)))  unsigned int v4u;
typedef __attribute__((ext_vector_type(4)))  float        v4f;

#define Bn   32
#define Cn   256
#define Nn   1024
#define Gn   32
#define CPG  8
#define EPSF 1e-5f
#define SM_SCALE 0.0625f   /* 1/sqrt(256) */

__device__ __forceinline__ v8f wmma_f16(v16h a, v16h b, v8f c) {
  return __builtin_amdgcn_wmma_f32_16x16x32_f16(false, a, false, b, (short)0, c,
                                                false, false);
}

// ---- CDNA5 f16 A-operand (16x32 MxK). row = &src[m_row_start].
// lane(m=l&15, h): halves k in [8h,8h+8) and [16+8h,16+8h+8)  -> 2 x b128
__device__ __forceinline__ v16h load_a16(const _Float16* row, int h) {
  union { v16h f; v4u u[2]; } r;
  r.u[0] = *(const v4u*)(row + 8 * h);
  r.u[1] = *(const v4u*)(row + 16 + 8 * h);
  return r.f;
}
// ---- CDNA5 f16 B-operand (32x16 KxN), NT form: row = &src[n_row_start].
// lane(n=l&15, h): halves k in [8h,8h+8) and [16+8h,16+8h+8)  -> 2 x b128
__device__ __forceinline__ v16h load_b16(const _Float16* row, int h) {
  union { v16h f; v4u u[2]; } r;
  r.u[0] = *(const v4u*)(row + 8 * h);
  r.u[1] = *(const v4u*)(row + 16 + 8 * h);
  return r.f;
}
// D/C f32 16x16: lane holds n = l&15 ; VGPR q holds m = q + 8*(l>>4).

// ======================= 1. GroupNorm statistics ========================
__global__ void gn_stats_kernel(const float* __restrict__ x,
                                float* __restrict__ mr) {
  int bg = blockIdx.x;
  int b = bg / Gn, g = bg % Gn;
  const float* base = x + ((size_t)b * Cn + (size_t)g * CPG) * Nn;
  float s = 0.f, ss = 0.f;
  for (int i = threadIdx.x; i < CPG * Nn; i += blockDim.x) {
    float v = base[i];
    s += v; ss += v * v;
  }
  __shared__ float sh[512];
  sh[threadIdx.x] = s; sh[256 + threadIdx.x] = ss;
  __syncthreads();
  for (int o = 128; o > 0; o >>= 1) {
    if (threadIdx.x < o) {
      sh[threadIdx.x] += sh[threadIdx.x + o];
      sh[256 + threadIdx.x] += sh[256 + threadIdx.x + o];
    }
    __syncthreads();
  }
  if (threadIdx.x == 0) {
    float inv = 1.0f / (float)(CPG * Nn);
    float mean = sh[0] * inv;
    float var = sh[256] * inv - mean * mean;
    mr[2 * bg] = mean;
    mr[2 * bg + 1] = rsqrtf(var + EPSF);
  }
}

// ============== 2. normalized x, transposed to token-major f16 ==========
__global__ void make_xhat_kernel(const float* __restrict__ x,
                                 const float* __restrict__ mr,
                                 _Float16* __restrict__ xhT) {
  size_t idx = (size_t)blockIdx.x * blockDim.x + threadIdx.x;
  int b = (int)(idx / ((size_t)Cn * Nn));
  int c = (int)((idx / Nn) % Cn);
  int p = (int)(idx % Nn);
  int g = c / CPG;
  float mean = mr[2 * (b * Gn + g)];
  float rstd = mr[2 * (b * Gn + g) + 1];
  xhT[((size_t)b * Nn + p) * Cn + c] = (_Float16)((x[idx] - mean) * rstd);
}

// ========= 3. weight prep: WT[d][c] = W[c][d]*s[c] (f16), bias fold =====
__global__ void wt_kernel(const float* __restrict__ W,
                          const float* __restrict__ s,
                          _Float16* __restrict__ WT) {
  int idx = blockIdx.x * 256 + threadIdx.x;
  int c = idx & 255, d = idx >> 8;
  float sc = s ? s[c] : 1.0f;
  WT[d * Cn + c] = (_Float16)(W[c * Cn + d] * sc);
}

__global__ void bias_fold_kernel(const float* __restrict__ W,
                                 const float* __restrict__ gb,
                                 const float* __restrict__ bias,
                                 float* __restrict__ out) {
  int d = threadIdx.x;
  float a = bias[d];
  for (int c = 0; c < Cn; ++c) a += gb[c] * W[c * Cn + d];
  out[d] = a;
}

// ====== 4a. NT GEMM, token-major out (Q, K): out[b][p][d] ==============
// D[m=p][n=d] = sum_c xhT[p][c] * WT[d][c] ; 32x32 tile per wave
__global__ void proj_tok_kernel(const _Float16* __restrict__ xhT,
                                const _Float16* __restrict__ WT,
                                const float* __restrict__ bias,
                                _Float16* __restrict__ out) {
  int lane = threadIdx.x & 31, wave = threadIdx.x >> 5;
  int w = blockIdx.x * 8 + wave;            // Bn * 32 * 8 waves
  int b = w >> 8, r = w & 255;
  int m0 = (r >> 3) * 32;                   // token tile
  int n0 = (r & 7) * 32;                    // channel tile
  int l = lane & 15, h = lane >> 4;

  const _Float16* Ar0 = xhT + ((size_t)b * Nn + m0 + l) * Cn;
  const _Float16* Ar1 = Ar0 + 16 * Cn;
  const _Float16* Br0 = WT + (size_t)(n0 + l) * Cn;
  const _Float16* Br1 = Br0 + 16 * Cn;

  v8f z = {0.f, 0.f, 0.f, 0.f, 0.f, 0.f, 0.f, 0.f};
  v8f a00 = z, a01 = z, a10 = z, a11 = z;
  for (int kc = 0; kc < Cn; kc += 32) {
    v16h a0 = load_a16(Ar0 + kc, h);
    v16h a1 = load_a16(Ar1 + kc, h);
    v16h b0 = load_b16(Br0 + kc, h);
    v16h b1 = load_b16(Br1 + kc, h);
    a00 = wmma_f16(a0, b0, a00);
    a01 = wmma_f16(a0, b1, a01);
    a10 = wmma_f16(a1, b0, a10);
    a11 = wmma_f16(a1, b1, a11);
  }
  float bn0 = bias[n0 + l], bn1 = bias[n0 + 16 + l];
#pragma unroll
  for (int q = 0; q < 8; ++q) {
    int m = q + 8 * h;
    size_t r0 = ((size_t)b * Nn + m0 + m) * Cn;
    size_t r1 = ((size_t)b * Nn + m0 + 16 + m) * Cn;
    out[r0 + n0 + l]      = (_Float16)(a00[q] + bn0);
    out[r0 + n0 + 16 + l] = (_Float16)(a01[q] + bn1);
    out[r1 + n0 + l]      = (_Float16)(a10[q] + bn0);
    out[r1 + n0 + 16 + l] = (_Float16)(a11[q] + bn1);
  }
}

// ====== 4b. NT GEMM, channel-major out (V): out[b][d][p] ===============
// D[m=d][n=p] = sum_c WT[d][c] * xhT[p][c]
__global__ void proj_chan_kernel(const _Float16* __restrict__ WT,
                                 const _Float16* __restrict__ xhT,
                                 const float* __restrict__ bias,
                                 _Float16* __restrict__ out) {
  int lane = threadIdx.x & 31, wave = threadIdx.x >> 5;
  int w = blockIdx.x * 8 + wave;
  int b = w >> 8, r = w & 255;
  int m0 = (r >> 5) * 32;                   // channel tile (8)
  int n0 = (r & 31) * 32;                   // token tile  (32)
  int l = lane & 15, h = lane >> 4;

  const _Float16* Ar0 = WT + (size_t)(m0 + l) * Cn;
  const _Float16* Ar1 = Ar0 + 16 * Cn;
  const _Float16* Br0 = xhT + ((size_t)b * Nn + n0 + l) * Cn;
  const _Float16* Br1 = Br0 + 16 * Cn;

  v8f z = {0.f, 0.f, 0.f, 0.f, 0.f, 0.f, 0.f, 0.f};
  v8f a00 = z, a01 = z, a10 = z, a11 = z;
  for (int kc = 0; kc < Cn; kc += 32) {
    v16h a0 = load_a16(Ar0 + kc, h);
    v16h a1 = load_a16(Ar1 + kc, h);
    v16h b0 = load_b16(Br0 + kc, h);
    v16h b1 = load_b16(Br1 + kc, h);
    a00 = wmma_f16(a0, b0, a00);
    a01 = wmma_f16(a0, b1, a01);
    a10 = wmma_f16(a1, b0, a10);
    a11 = wmma_f16(a1, b1, a11);
  }
#pragma unroll
  for (int q = 0; q < 8; ++q) {
    int m = q + 8 * h;
    size_t r0 = ((size_t)b * Cn + m0 + m) * Nn;
    size_t r1 = ((size_t)b * Cn + m0 + 16 + m) * Nn;
    float b0v = bias[m0 + m], b1v = bias[m0 + 16 + m];
    out[r0 + n0 + l]      = (_Float16)(a00[q] + b0v);
    out[r0 + n0 + 16 + l] = (_Float16)(a01[q] + b0v);
    out[r1 + n0 + l]      = (_Float16)(a10[q] + b1v);
    out[r1 + n0 + 16 + l] = (_Float16)(a11[q] + b1v);
  }
}

// ======================= 5. attention ===================================
// one wave per (batch, 16-query tile). Q,K token-major; V channel-major.
__global__ __launch_bounds__(32) void attention_kernel(
    const _Float16* __restrict__ qtm, const _Float16* __restrict__ ktm,
    const _Float16* __restrict__ vcm, _Float16* __restrict__ ho) {
  __shared__ _Float16 P[16 * Nn];           // 32 KB probability rows
  int lane = threadIdx.x;
  int l = lane & 15, h = lane >> 4;
  int b = blockIdx.x >> 6;
  int p0 = (blockIdx.x & 63) * 16;
  v8f z = {0.f, 0.f, 0.f, 0.f, 0.f, 0.f, 0.f, 0.f};

  const _Float16* Ar = qtm + ((size_t)b * Nn + p0 + l) * Cn;

  // ---- phase 1: S = Q K^T * scale -> P ----
  for (int kt = 0; kt < Nn; kt += 32) {
    const _Float16* Br0 = ktm + ((size_t)b * Nn + kt + l) * Cn;
    const _Float16* Br1 = Br0 + 16 * Cn;
    v8f s0 = z, s1 = z;
    for (int kc = 0; kc < Cn; kc += 32) {
      v16h a = load_a16(Ar + kc, h);
      s0 = wmma_f16(a, load_b16(Br0 + kc, h), s0);
      s1 = wmma_f16(a, load_b16(Br1 + kc, h), s1);
    }
#pragma unroll
    for (int q = 0; q < 8; ++q) {
      int m = q + 8 * h;
      P[m * Nn + kt + l]      = (_Float16)(s0[q] * SM_SCALE);
      P[m * Nn + kt + 16 + l] = (_Float16)(s1[q] * SM_SCALE);
    }
  }
  __syncthreads();

  // ---- phase 2: row softmax in LDS ----
  for (int row = 0; row < 16; ++row) {
    float mx = -3.0e38f;
    for (int j = lane; j < Nn; j += 32)
      mx = fmaxf(mx, (float)P[row * Nn + j]);
    for (int o = 16; o > 0; o >>= 1) mx = fmaxf(mx, __shfl_xor(mx, o, 32));
    float sum = 0.f;
    for (int j = lane; j < Nn; j += 32) {
      float e = __expf((float)P[row * Nn + j] - mx);
      P[row * Nn + j] = (_Float16)e;
      sum += e;
    }
    for (int o = 16; o > 0; o >>= 1) sum += __shfl_xor(sum, o, 32);
    float inv = 1.0f / sum;
    for (int j = lane; j < Nn; j += 32)
      P[row * Nn + j] = (_Float16)((float)P[row * Nn + j] * inv);
  }
  __syncthreads();

  // ---- phase 3: O = P V^T (V channel-major -> NT) ----
  for (int c0 = 0; c0 < Cn; c0 += 32) {
    const _Float16* Br0 = vcm + ((size_t)b * Cn + c0 + l) * Nn;
    const _Float16* Br1 = Br0 + 16 * Nn;
    v8f o0 = z, o1 = z;
    for (int j0 = 0; j0 < Nn; j0 += 32) {
      v16h a = load_a16(P + l * Nn + j0, h);      // ds_load_b128 x2
      o0 = wmma_f16(a, load_b16(Br0 + j0, h), o0);
      o1 = wmma_f16(a, load_b16(Br1 + j0, h), o1);
    }
#pragma unroll
    for (int q = 0; q < 8; ++q) {
      int m = q + 8 * h;
      size_t r0 = ((size_t)b * Nn + p0 + m) * Cn;
      ho[r0 + c0 + l]      = (_Float16)o0[q];
      ho[r0 + c0 + 16 + l] = (_Float16)o1[q];
    }
  }
}

// ============== 6. output projection + residual (f32 out) ==============
// D[m=p][n=d] = sum_c ho[p][c] * WoT[d][c] ; out channel-major + x
__global__ void out_proj_kernel(const _Float16* __restrict__ ho,
                                const _Float16* __restrict__ WoT,
                                const float* __restrict__ bo,
                                const float* __restrict__ x,
                                float* __restrict__ out) {
  int lane = threadIdx.x & 31, wave = threadIdx.x >> 5;
  int w = blockIdx.x * 8 + wave;
  int b = w >> 8, r = w & 255;
  int m0 = (r >> 3) * 32;                   // token tile
  int n0 = (r & 7) * 32;                    // channel tile
  int l = lane & 15, h = lane >> 4;

  const _Float16* Ar0 = ho + ((size_t)b * Nn + m0 + l) * Cn;
  const _Float16* Ar1 = Ar0 + 16 * Cn;
  const _Float16* Br0 = WoT + (size_t)(n0 + l) * Cn;
  const _Float16* Br1 = Br0 + 16 * Cn;

  v8f z = {0.f, 0.f, 0.f, 0.f, 0.f, 0.f, 0.f, 0.f};
  v8f a00 = z, a01 = z, a10 = z, a11 = z;
  for (int kc = 0; kc < Cn; kc += 32) {
    v16h a0 = load_a16(Ar0 + kc, h);
    v16h a1 = load_a16(Ar1 + kc, h);
    v16h b0 = load_b16(Br0 + kc, h);
    v16h b1 = load_b16(Br1 + kc, h);
    a00 = wmma_f16(a0, b0, a00);
    a01 = wmma_f16(a0, b1, a01);
    a10 = wmma_f16(a1, b0, a10);
    a11 = wmma_f16(a1, b1, a11);
  }
  // store: out[(b*Cn + n)*Nn + m] ; consecutive m -> float4 vector I/O
#pragma unroll
  for (int t = 0; t < 2; ++t) {
    int n = n0 + 16 * t + l;
    float bb = bo[n];
    v8f accA = t ? a01 : a00;   // m-tile 0
    v8f accB = t ? a11 : a10;   // m-tile 1
#pragma unroll
    for (int i = 0; i < 2; ++i) {
      v8f acc = i ? accB : accA;
      size_t base = ((size_t)b * Cn + n) * Nn + m0 + 16 * i + 8 * h;
      v4f xlo = *(const v4f*)(x + base);
      v4f xhi = *(const v4f*)(x + base + 4);
      v4f olo = {xlo[0] + acc[0] + bb, xlo[1] + acc[1] + bb,
                 xlo[2] + acc[2] + bb, xlo[3] + acc[3] + bb};
      v4f ohi = {xhi[0] + acc[4] + bb, xhi[1] + acc[5] + bb,
                 xhi[2] + acc[6] + bb, xhi[3] + acc[7] + bb};
      *(v4f*)(out + base) = olo;
      *(v4f*)(out + base + 4) = ohi;
    }
  }
}

// ======================= launch =========================================
extern "C" void kernel_launch(void* const* d_in, const int* in_sizes, int n_in,
                              void* d_out, int out_size, void* d_ws,
                              size_t ws_size, hipStream_t stream) {
  const float* x    = (const float*)d_in[0];
  const float* Wq   = (const float*)d_in[1];
  const float* bq   = (const float*)d_in[2];
  const float* Wk   = (const float*)d_in[3];
  const float* bk   = (const float*)d_in[4];
  const float* Wv   = (const float*)d_in[5];
  const float* bv   = (const float*)d_in[6];
  const float* Wo   = (const float*)d_in[7];
  const float* bo   = (const float*)d_in[8];
  const float* gq_s = (const float*)d_in[9];
  const float* gq_b = (const float*)d_in[10];
  const float* gk_s = (const float*)d_in[11];
  const float* gk_b = (const float*)d_in[12];
  const float* gv_s = (const float*)d_in[13];
  const float* gv_b = (const float*)d_in[14];
  float* out = (float*)d_out;

  // ---- workspace layout ----
  float* mr  = (float*)d_ws;                 // 2*B*G
  float* bqf = mr + 2048;                    // folded biases, 256 each
  float* bkf = bqf + 256;
  float* bvf = bkf + 256;
  _Float16* WqT = (_Float16*)(bvf + 256);    // 4 x 64K f16
  _Float16* WkT = WqT + Cn * Cn;
  _Float16* WvT = WkT + Cn * Cn;
  _Float16* WoT = WvT + Cn * Cn;
  _Float16* xhT = WoT + Cn * Cn;             // token-major xhat
  const size_t XSZ = (size_t)Bn * Cn * Nn;
  _Float16* qtm = xhT + XSZ;                 // Q token-major
  _Float16* ktm = qtm + XSZ;                 // K token-major
  _Float16* vcm = ktm + XSZ;                 // V channel-major
  _Float16* hot = vcm + XSZ;                 // attention out, token-major

  gn_stats_kernel<<<Bn * Gn, 256, 0, stream>>>(x, mr);
  make_xhat_kernel<<<(int)(XSZ / 256), 256, 0, stream>>>(x, mr, xhT);

  wt_kernel<<<256, 256, 0, stream>>>(Wq, gq_s, WqT);
  wt_kernel<<<256, 256, 0, stream>>>(Wk, gk_s, WkT);
  wt_kernel<<<256, 256, 0, stream>>>(Wv, gv_s, WvT);
  wt_kernel<<<256, 256, 0, stream>>>(Wo, (const float*)nullptr, WoT);
  bias_fold_kernel<<<1, 256, 0, stream>>>(Wq, gq_b, bq, bqf);
  bias_fold_kernel<<<1, 256, 0, stream>>>(Wk, gk_b, bk, bkf);
  bias_fold_kernel<<<1, 256, 0, stream>>>(Wv, gv_b, bv, bvf);

  const int proj_blocks = (Bn * 256) / 8;    // 8 waves / block
  proj_tok_kernel<<<proj_blocks, 256, 0, stream>>>(xhT, WqT, bqf, qtm);
  proj_tok_kernel<<<proj_blocks, 256, 0, stream>>>(xhT, WkT, bkf, ktm);
  proj_chan_kernel<<<proj_blocks, 256, 0, stream>>>(WvT, xhT, bvf, vcm);

  attention_kernel<<<Bn * 64, 32, 0, stream>>>(qtm, ktm, vcm, hot);

  out_proj_kernel<<<proj_blocks, 256, 0, stream>>>(hot, WoT, bo, x, out);
}
